// SetTransEncoder_53094385713790
// MI455X (gfx1250) — matrix operand order repaired and checked by reading, not compiled, not run on hardware
//
#include <hip/hip_runtime.h>
#include <stdint.h>

#define B_   64
#define N_   2048
#define M_   32
#define D_   128
#define H_   4
#define DH_  32
#define HDH_ 128   // H*DH
#define DFF_ 512
#define L_   2

typedef unsigned int   u32;
typedef unsigned short u16;
typedef __attribute__((ext_vector_type(16))) __bf16 v16bf;
typedef __attribute__((ext_vector_type(8)))  float  v8f;
typedef __attribute__((ext_vector_type(8)))  u32    v8u;

// ---------- bf16 helpers ----------
__device__ __forceinline__ u16 f2bf(float f) {
  u32 u = __builtin_bit_cast(u32, f);
  u32 r = u + 0x7fffu + ((u >> 16) & 1u);   // round-to-nearest-even
  return (u16)(r >> 16);
}
__device__ __forceinline__ float bf2f(u16 h) {
  u32 u = ((u32)h) << 16;
  return __builtin_bit_cast(float, u);
}
__device__ __forceinline__ u32 pack2bf(float lo, float hi) {
  return (u32)f2bf(lo) | ((u32)f2bf(hi) << 16);
}
__device__ __forceinline__ v8f vzero8() {
  v8f z;
  #pragma unroll
  for (int i = 0; i < 8; ++i) z[i] = 0.0f;
  return z;
}
__device__ __forceinline__ v16bf frag_from(uint4 lo, uint4 hi) {
  v8u u;
  u[0] = lo.x; u[1] = lo.y; u[2] = lo.z; u[3] = lo.w;
  u[4] = hi.x; u[5] = hi.y; u[6] = hi.z; u[7] = hi.w;
  return __builtin_bit_cast(v16bf, u);
}
__device__ __forceinline__ v8f wmma_bf16(v16bf a, v16bf b, v8f c) {
  return __builtin_amdgcn_wmma_f32_16x16x32_bf16(false, a, false, b, (short)0, c,
                                                 false, false);
}

// ---------- fragment loaders (k = 16*half + s in both A and B) ----------
// A 16x32 bf16 from row-major src (stride ld elems): two contiguous 16B chunks.
__device__ __forceinline__ v16bf load_a_frag(const u16* src, int ld, int lane) {
  const int m = lane & 15, half = lane >> 4;
  const u16* p = src + (size_t)m * ld + 16 * half;
  uint4 lo = *(const uint4*)p;
  uint4 hi = *(const uint4*)(p + 8);
  return frag_from(lo, hi);
}
// B 32x16 bf16 from N-major storage src[n][k] (stride ld): 16 contiguous bf16.
__device__ __forceinline__ v16bf load_b_frag_nk(const u16* src, int ld, int lane) {
  const int n = lane & 15, half = lane >> 4;
  const u16* p = src + (size_t)n * ld + 16 * half;
  uint4 lo = *(const uint4*)p;
  uint4 hi = *(const uint4*)(p + 8);
  return frag_from(lo, hi);
}
// B fragment from pre-packed weights: Wp[tile][lane][16 bf16] (32B per lane).
__device__ __forceinline__ v16bf load_b_frag_packed(const u16* Wp, size_t tile,
                                                    int lane) {
  const u16* p = Wp + (tile * 32 + lane) * 16;
  uint4 lo = *(const uint4*)p;
  uint4 hi = *(const uint4*)(p + 8);
  return frag_from(lo, hi);
}

// ---------- weight packing: W[K][N] f32 -> fragment-native bf16 ----------
__global__ void pack_w_kernel(const float* __restrict__ W, u16* __restrict__ Wp,
                              int K, int N, int total) {
  int i = blockIdx.x * 256 + threadIdx.x;
  if (i >= total) return;
  const int per = K * N;
  int s = i / per;
  int r = i % per;
  int e = r & 15;
  int lane = (r >> 4) & 31;
  int t = r >> 9;
  int ntiles = N >> 4;
  int nt = t % ntiles;
  int kt = t / ntiles;
  int k = kt * 32 + 2 * (e >> 1) + (e & 1) + 16 * (lane >> 4);
  int n = nt * 16 + (lane & 15);
  Wp[i] = f2bf(W[(size_t)s * per + (size_t)k * N + n]);
}

// ---------- generic GEMM: C[rows,N] = act(A[rows,K] @ Wp + bias) ----------
// FLAGS (compile-time): 1 = A is bf16, 2 = relu, 4 = store bf16, 8 = add bias
// Block = 128 threads (4 waves). Tile: 32 rows x 128 cols. Wave w owns
// n-tiles w*4..w*4+3 across both 16-row m-tiles (8 WMMA accumulators).
template <int FLAGS>
__global__ void __launch_bounds__(128) gemm_bf16_kernel(
    const void* __restrict__ Av, const u16* __restrict__ Wp,
    const float* __restrict__ bias, void* __restrict__ Cv,
    int rows, int K, int N) {
  __shared__ __align__(16) u16 ldsA[32 * 32];
  const int tid = threadIdx.x;
  const int lane = tid & 31;
  const int wave = tid >> 5;
  const int row0 = blockIdx.x * 32;
  const int nblk = blockIdx.y;
  const float* Af = (const float*)Av;
  const u16*  Ab = (const u16*)Av;
  const int ntiles_total = N >> 4;

  v8f acc[2][4];
  #pragma unroll
  for (int mt = 0; mt < 2; ++mt)
    #pragma unroll
    for (int nt = 0; nt < 4; ++nt) acc[mt][nt] = vzero8();

  const int sr = tid >> 2;              // staging row 0..31
  const int sc = (tid & 3) * 8;         // staging col 0,8,16,24

  for (int k0 = 0; k0 < K; k0 += 32) {
    __syncthreads();
    if (FLAGS & 1) {
      uint4 v = *(const uint4*)(Ab + (size_t)(row0 + sr) * K + k0 + sc);
      *(uint4*)(ldsA + sr * 32 + sc) = v;
    } else {
      const float* ap = Af + (size_t)(row0 + sr) * K + k0 + sc;
      float4 f0 = *(const float4*)ap;
      float4 f1 = *(const float4*)(ap + 4);
      uint4 v;
      v.x = pack2bf(f0.x, f0.y); v.y = pack2bf(f0.z, f0.w);
      v.z = pack2bf(f1.x, f1.y); v.w = pack2bf(f1.z, f1.w);
      *(uint4*)(ldsA + sr * 32 + sc) = v;
    }
    __syncthreads();
    v16bf af0 = load_a_frag(ldsA, 32, lane);
    v16bf af1 = load_a_frag(ldsA + 16 * 32, 32, lane);
    const size_t ktile = (size_t)(k0 >> 5) * ntiles_total;
    #pragma unroll
    for (int nt = 0; nt < 4; ++nt) {
      size_t gnt = (size_t)(nblk * 8 + wave * 4 + nt);
      v16bf bf = load_b_frag_packed(Wp, ktile + gnt, lane);
      acc[0][nt] = wmma_bf16(af0, bf, acc[0][nt]);
      acc[1][nt] = wmma_bf16(af1, bf, acc[1][nt]);
    }
  }

  const int half = lane >> 4;
  const int nlo  = lane & 15;
  #pragma unroll
  for (int mt = 0; mt < 2; ++mt) {
    #pragma unroll
    for (int nt = 0; nt < 4; ++nt) {
      int col = nblk * 128 + (wave * 4 + nt) * 16 + nlo;
      float bv = (FLAGS & 8) ? bias[col] : 0.0f;
      #pragma unroll
      for (int r = 0; r < 8; ++r) {
        int row = row0 + mt * 16 + r + 8 * half;
        float v = acc[mt][nt][r] + bv;
        if (FLAGS & 2) v = fmaxf(v, 0.0f);
        size_t g = (size_t)row * N + col;
        if (FLAGS & 4) ((u16*)Cv)[g] = f2bf(v);
        else           ((float*)Cv)[g] = v;
      }
    }
  }
}

// ---------- attention scores: S[b,h,q,k] = scale * Q·K^T ----------
__global__ void __launch_bounds__(128) attn_scores_kernel(
    const u16* __restrict__ Qp, const u16* __restrict__ Kp,
    float* __restrict__ S, int nq, int nk, float scale) {
  int gw = (blockIdx.x * 128 + threadIdx.x) >> 5;
  const int lane = threadIdx.x & 31;
  const int tk = nk >> 4, tq = nq >> 4;
  int kt = gw % tk; gw /= tk;
  int qt = gw % tq; gw /= tq;
  int h = gw % H_;  int b = gw / H_;
  if (b >= B_) return;  // wave-uniform
  const u16* qb = Qp + (size_t)(b * nq + qt * 16) * HDH_ + h * DH_;
  const u16* kb = Kp + (size_t)(b * nk + kt * 16) * HDH_ + h * DH_;
  v16bf a  = load_a_frag(qb, HDH_, lane);
  v16bf bf = load_b_frag_nk(kb, HDH_, lane);   // K rows are the N-major B matrix
  v8f c = vzero8();
  c = wmma_bf16(a, bf, c);
  const int half = lane >> 4, n = lane & 15;
  size_t base = ((size_t)(b * H_ + h) * nq + qt * 16) * nk + kt * 16 + n;
  #pragma unroll
  for (int r = 0; r < 8; ++r)
    S[base + (size_t)(r + 8 * half) * nk] = c[r] * scale;
}

// ---------- row softmax, fp32 in -> bf16 probs out (wave per row) ----------
__global__ void __launch_bounds__(128) softmax_rows_kernel(
    const float* __restrict__ S, u16* __restrict__ P, int nk) {
  const int row  = (blockIdx.x * 128 + threadIdx.x) >> 5;
  const int lane = threadIdx.x & 31;
  const float* srow = S + (size_t)row * nk;
  u16* pr = P + (size_t)row * nk;
  const int per = nk >> 5;
  float mx = -3.0e38f;
  for (int i = 0; i < per; ++i) mx = fmaxf(mx, srow[lane + 32 * i]);
  #pragma unroll
  for (int o = 16; o >= 1; o >>= 1) mx = fmaxf(mx, __shfl_xor(mx, o, 32));
  float sum = 0.f;
  for (int i = 0; i < per; ++i) {
    float e = __expf(srow[lane + 32 * i] - mx);
    sum += e;
    pr[lane + 32 * i] = f2bf(e);
  }
  #pragma unroll
  for (int o = 16; o >= 1; o >>= 1) sum += __shfl_xor(sum, o, 32);
  float inv = 1.0f / sum;
  for (int i = 0; i < per; ++i)
    pr[lane + 32 * i] = f2bf(bf2f(pr[lane + 32 * i]) * inv);
}

// ---------- V transpose: Vp[b*nk+key][h*32+d] -> Vt[b][h*32+d][key] ----------
__global__ void transpose_v_kernel(const u16* __restrict__ Vp,
                                   u16* __restrict__ Vt, int nk) {
  int i = blockIdx.x * 256 + threadIdx.x;
  if (i >= B_ * nk * HDH_) return;
  int d = i % HDH_;
  int key = (i / HDH_) % nk;
  int b = i / (HDH_ * nk);
  Vt[((size_t)b * HDH_ + d) * nk + key] = Vp[i];
}

// ---------- att @ V: O[b,q,h*32+d] f32 ----------
__global__ void __launch_bounds__(128) attn_av_kernel(
    const u16* __restrict__ P, const u16* __restrict__ Vt,
    float* __restrict__ O, int nq, int nk) {
  int gw = (blockIdx.x * 128 + threadIdx.x) >> 5;
  const int lane = threadIdx.x & 31;
  int dt = gw & 1; gw >>= 1;
  int h  = gw % H_; gw /= H_;
  const int tq = nq >> 4;
  int qt = gw % tq; int b = gw / tq;
  if (b >= B_) return;  // wave-uniform
  const int half = lane >> 4, n = lane & 15;
  const int col = h * DH_ + dt * 16 + n;
  const u16* prow = P + ((size_t)(b * H_ + h) * nq + qt * 16) * nk;
  const u16* vrow = Vt + (size_t)(b * HDH_ + col) * nk + 16 * half;
  v8f acc = vzero8();
  for (int kt = 0; kt < (nk >> 5); ++kt) {
    v16bf a = load_a_frag(prow + kt * 32, nk, lane);
    const u16* vp = vrow + kt * 32;
    uint4 lo = *(const uint4*)vp;
    uint4 hi = *(const uint4*)(vp + 8);
    acc = wmma_bf16(a, frag_from(lo, hi), acc);
  }
  size_t rbase = (size_t)(b * nq + qt * 16 + 8 * half) * HDH_ + col;
  #pragma unroll
  for (int r = 0; r < 8; ++r)
    O[rbase + (size_t)r * HDH_] = acc[r];
}

// ---------- residual + LayerNorm over D=128 (wave per row) ----------
__global__ void __launch_bounds__(256) ln_res_kernel(
    const float* __restrict__ a, const float* __restrict__ b,
    const float* __restrict__ g, const float* __restrict__ bet,
    float* __restrict__ out) {
  const int row  = (blockIdx.x * 256 + threadIdx.x) >> 5;
  const int lane = threadIdx.x & 31;
  const float* ar = a + (size_t)row * D_;
  const float* br = b + (size_t)row * D_;
  float x[4]; float s = 0.f;
  #pragma unroll
  for (int i = 0; i < 4; ++i) { x[i] = ar[lane + 32 * i] + br[lane + 32 * i]; s += x[i]; }
  #pragma unroll
  for (int o = 16; o >= 1; o >>= 1) s += __shfl_xor(s, o, 32);
  const float mu = s * (1.0f / D_);
  float vs = 0.f;
  #pragma unroll
  for (int i = 0; i < 4; ++i) { float d = x[i] - mu; vs += d * d; }
  #pragma unroll
  for (int o = 16; o >= 1; o >>= 1) vs += __shfl_xor(vs, o, 32);
  const float inv = rsqrtf(vs * (1.0f / D_) + 1e-5f);
  float* orow = out + (size_t)row * D_;
  #pragma unroll
  for (int i = 0; i < 4; ++i) {
    int c = lane + 32 * i;
    orow[c] = (x[i] - mu) * inv * g[c] + bet[c];
  }
}

// ---------- broadcast inducing points over batch ----------
__global__ void bcast_rows_kernel(const float* __restrict__ s, float* __restrict__ d,
                                  int rep, int n) {
  int i = blockIdx.x * 256 + threadIdx.x;
  if (i < rep * n) d[i] = s[i % n];
}

extern "C" void kernel_launch(void* const* d_in, const int* in_sizes, int n_in,
                              void* d_out, int out_size, void* d_ws, size_t ws_size,
                              hipStream_t stream) {
  (void)in_sizes; (void)n_in; (void)out_size; (void)ws_size;
  const float* feat = (const float*)d_in[0];
  const float* Ipts = (const float*)d_in[1];
  const float* Wq   = (const float*)d_in[2];
  const float* Wk   = (const float*)d_in[3];
  const float* Wv   = (const float*)d_in[4];
  const float* Wo   = (const float*)d_in[5];
  const float* W1   = (const float*)d_in[6];
  const float* b1   = (const float*)d_in[7];
  const float* W2   = (const float*)d_in[8];
  const float* b2   = (const float*)d_in[9];
  const float* ln1g = (const float*)d_in[10];
  const float* ln1b = (const float*)d_in[11];
  const float* ln2g = (const float*)d_in[12];
  const float* ln2b = (const float*)d_in[13];

  char* ws = (char*)d_ws;
  size_t off = 0;
  auto carve = [&](size_t bytes) -> char* {
    char* p = ws + off;
    off = (off + bytes + 255) & ~(size_t)255;
    return p;
  };
  const int NPROJ = D_ * HDH_;            // 16384 per (l,j)
  const int NFFN  = D_ * DFF_;            // 65536 per (l,j)
  u16*  wq_b = (u16*)carve(sizeof(u16) * L_ * 2 * NPROJ);
  u16*  wk_b = (u16*)carve(sizeof(u16) * L_ * 2 * NPROJ);
  u16*  wv_b = (u16*)carve(sizeof(u16) * L_ * 2 * NPROJ);
  u16*  wo_b = (u16*)carve(sizeof(u16) * L_ * 2 * NPROJ);
  u16*  w1_b = (u16*)carve(sizeof(u16) * L_ * 2 * NFFN);
  u16*  w2_b = (u16*)carve(sizeof(u16) * L_ * 2 * NFFN);
  u16*  Qp   = (u16*)carve(sizeof(u16) * (size_t)B_ * N_ * HDH_);
  u16*  Kp   = (u16*)carve(sizeof(u16) * (size_t)B_ * N_ * HDH_);
  u16*  Vp   = (u16*)carve(sizeof(u16) * (size_t)B_ * N_ * HDH_);
  u16*  Vt   = (u16*)carve(sizeof(u16) * (size_t)B_ * N_ * HDH_);
  float* S   = (float*)carve(sizeof(float) * (size_t)B_ * H_ * N_ * M_);
  u16*  P    = (u16*)carve(sizeof(u16) * (size_t)B_ * H_ * N_ * M_);
  float* O   = (float*)carve(sizeof(float) * (size_t)B_ * N_ * HDH_);
  float* SA  = (float*)carve(sizeof(float) * (size_t)B_ * N_ * D_);  // also FFN out
  float* X1  = (float*)carve(sizeof(float) * (size_t)B_ * N_ * D_);
  u16*  Hb   = (u16*)carve(sizeof(u16) * (size_t)B_ * N_ * DFF_);
  float* xb  = (float*)carve(sizeof(float) * (size_t)B_ * N_ * D_);
  float* hq  = (float*)carve(sizeof(float) * (size_t)B_ * M_ * D_);
  float* hb  = (float*)carve(sizeof(float) * (size_t)B_ * M_ * D_);

  auto packw = [&](const float* s, u16* d, int K, int N) {
    int total = L_ * 2 * K * N;
    pack_w_kernel<<<(total + 255) / 256, 256, 0, stream>>>(s, d, K, N, total);
  };
  packw(Wq, wq_b, D_, HDH_);
  packw(Wk, wk_b, D_, HDH_);
  packw(Wv, wv_b, D_, HDH_);
  packw(Wo, wo_b, HDH_, D_);
  packw(W1, w1_b, D_, DFF_);
  packw(W2, w2_b, DFF_, D_);

  auto mha = [&](const float* q_src, int nq, const float* kv_src, int nk,
                 int idx, float* outp) {
    const int rows_q = B_ * nq, rows_k = B_ * nk;
    // Q/K/V projections -> bf16 outputs (FLAGS=4)
    gemm_bf16_kernel<4><<<dim3(rows_q / 32, 1), 128, 0, stream>>>(
        q_src, wq_b + (size_t)idx * NPROJ, nullptr, Qp, rows_q, D_, HDH_);
    gemm_bf16_kernel<4><<<dim3(rows_k / 32, 1), 128, 0, stream>>>(
        kv_src, wk_b + (size_t)idx * NPROJ, nullptr, Kp, rows_k, D_, HDH_);
    gemm_bf16_kernel<4><<<dim3(rows_k / 32, 1), 128, 0, stream>>>(
        kv_src, wv_b + (size_t)idx * NPROJ, nullptr, Vp, rows_k, D_, HDH_);
    transpose_v_kernel<<<(B_ * nk * HDH_ + 255) / 256, 256, 0, stream>>>(Vp, Vt, nk);
    const float scale = 0.17677669529663687f;  // 1/sqrt(32)
    int st = B_ * H_ * (nq / 16) * (nk / 16);
    attn_scores_kernel<<<st / 4, 128, 0, stream>>>(Qp, Kp, S, nq, nk, scale);
    int srows = B_ * H_ * nq;
    softmax_rows_kernel<<<srows / 4, 128, 0, stream>>>(S, P, nk);
    int at = B_ * (nq / 16) * H_ * 2;
    attn_av_kernel<<<at / 4, 128, 0, stream>>>(P, Vt, O, nq, nk);
    // Wo: f32 out (FLAGS=0)
    gemm_bf16_kernel<0><<<dim3(rows_q / 32, 1), 128, 0, stream>>>(
        O, wo_b + (size_t)idx * NPROJ, nullptr, SA, rows_q, HDH_, D_);
    ln_res_kernel<<<rows_q / 8, 256, 0, stream>>>(q_src, SA,
        ln1g + (size_t)idx * D_, ln1b + (size_t)idx * D_, X1);
    // FFN1: relu + bias + bf16 out (FLAGS=2|4|8=14), N=512 -> grid.y=4
    gemm_bf16_kernel<14><<<dim3(rows_q / 32, DFF_ / 128), 128, 0, stream>>>(
        X1, w1_b + (size_t)idx * NFFN, b1 + (size_t)idx * DFF_, Hb,
        rows_q, D_, DFF_);
    // FFN2: bf16 A + bias + f32 out (FLAGS=1|8=9)
    gemm_bf16_kernel<9><<<dim3(rows_q / 32, 1), 128, 0, stream>>>(
        Hb, w2_b + (size_t)idx * NFFN, b2 + (size_t)idx * D_, SA,
        rows_q, DFF_, D_);
    ln_res_kernel<<<rows_q / 8, 256, 0, stream>>>(X1, SA,
        ln2g + (size_t)idx * D_, ln2b + (size_t)idx * D_, outp);
  };

  const float* x = feat;
  for (int l = 0; l < L_; ++l) {
    bcast_rows_kernel<<<(B_ * M_ * D_) / 256, 256, 0, stream>>>(
        Ipts + (size_t)l * M_ * D_, hq, B_, M_ * D_);
    mha(hq, M_, x, N_, l * 2 + 0, hb);                       // H = MAB(I, X)
    float* dst = (l == L_ - 1) ? (float*)d_out : xb;
    mha(x, N_, hb, M_, l * 2 + 1, dst);                      // X' = MAB(X, H)
    x = xb;
  }
}
